// MultiHeadAttention_50491635532194
// MI455X (gfx1250) — matrix operand order repaired and checked by reading, not compiled, not run on hardware
//
#include <hip/hip_runtime.h>

// ---------------------------------------------------------------------------
// MI455X (gfx1250) fused multi-head attention, fp32 in/out, f16 WMMA compute.
// B=8, S=2048, D=256, H=4, HD=64
// ---------------------------------------------------------------------------

typedef _Float16 half_t;
typedef __attribute__((ext_vector_type(8)))  _Float16 v8h;
typedef __attribute__((ext_vector_type(16))) _Float16 v16h;
typedef __attribute__((ext_vector_type(8)))  float    v8f;

#define BATCH 8
#define SEQ   2048
#define DMODEL 256
#define NHEAD 4
#define HDIM  64
#define ROWS (BATCH * SEQ)          // 16384

// ---------------- WMMA fragment helpers (wave32) ----------------------------
// A 16x32 f16: lane (l&15) = row M; half hi=(l>>4) holds K = hi*8..hi*8+7 and
// K = 16+hi*8..16+hi*8+7 (ISA 7.12.2, 16-bit A layout).
__device__ __forceinline__ v16h load_frag_a(const half_t* rowptr, int hi) {
  union { v16h v; v8h h[2]; } u;
  u.h[0] = *(const v8h*)(rowptr + hi * 8);
  u.h[1] = *(const v8h*)(rowptr + 16 + hi * 8);
  return u.v;
}
// B 32x16 f16 stored transposed in LDS as [N][K]: lane (l&15) = col N; half hi
// holds K = hi*8..hi*8+7 and K = 16+hi*8..16+hi*8+7 (same order as A).
__device__ __forceinline__ v16h load_frag_b(const half_t* colptr, int hi) {
  union { v16h v; v8h h[2]; } u;
  u.h[0] = *(const v8h*)(colptr + hi * 8);
  u.h[1] = *(const v8h*)(colptr + 16 + hi * 8);
  return u.v;
}
__device__ __forceinline__ v8f wmma32(v16h a, v16h b, v8f c) {
  return __builtin_amdgcn_wmma_f32_16x16x32_f16(
      /*neg_a=*/false, a, /*neg_b=*/false, b,
      /*c_mod=*/(short)0, c, /*reuse_a=*/false, /*reuse_b=*/false);
}

// ---------------------------------------------------------------------------
// Kernel 1: fused projection GEMM  out_f16[ROWS,256] = f16(x @ W + bias)
// 256 threads = 8 waves; block tile 128x64; K staged in 32-wide LDS slabs.
// ---------------------------------------------------------------------------
__global__ __launch_bounds__(256) void qkv_proj_kernel(
    const float* __restrict__ x, const float* __restrict__ W,
    const float* __restrict__ bias, half_t* __restrict__ out) {
  constexpr int TM = 128, TN = 64, TK = 32;
  __shared__ alignas(16) half_t lA[TM][TK];   // x tile, row-major
  __shared__ alignas(16) half_t lB[TN][TK];   // W tile, stored transposed [n][k]

  const int tid  = threadIdx.x;
  const int wave = tid >> 5;
  const int lane = tid & 31;
  const int hi   = lane >> 4;
  const int ln   = lane & 15;
  const int row0 = blockIdx.x * TM;
  const int n0   = blockIdx.y * TN;

  v8f acc[4] = {{}, {}, {}, {}};

  for (int kt = 0; kt < DMODEL; kt += TK) {
    // Stage A: 128x32 fp32 -> f16 (16 contiguous elements per thread).
    {
      const int base = tid * 16;
      const int r = base >> 5, c = base & 31;
      const float* src = x + (size_t)(row0 + r) * DMODEL + kt + c;
      half_t* dst = &lA[r][c];
#pragma unroll
      for (int i = 0; i < 16; ++i) dst[i] = (half_t)src[i];
    }
    // Stage B transposed: lB[n][k] = W[(kt+k)*D + n0+n]; coalesced reads.
    {
      const int c = tid >> 3;            // k within slab (0..31)
      const int n = (tid & 7) * 8;       // 8 consecutive n per thread
      const float* src = W + (size_t)(kt + c) * DMODEL + n0 + n;
#pragma unroll
      for (int i = 0; i < 8; ++i) lB[n + i][c] = (half_t)src[i];
    }
    __syncthreads();

    const v16h a = load_frag_a(&lA[wave * 16 + ln][0], hi);
#pragma unroll
    for (int j = 0; j < 4; ++j) {
      const v16h b = load_frag_b(&lB[j * 16 + ln][0], hi);
      acc[j] = wmma32(a, b, acc[j]);
    }
    __syncthreads();
  }

#pragma unroll
  for (int j = 0; j < 4; ++j) {
    const int col = n0 + j * 16 + ln;
    const float bv = bias[col];
#pragma unroll
    for (int r = 0; r < 8; ++r) {
      const int row = row0 + wave * 16 + r + 8 * hi;
      out[(size_t)row * DMODEL + col] = (half_t)(acc[j][r] + bv);
    }
  }
}

// ---------------------------------------------------------------------------
// Kernel 2: flash attention. grid (S/64, H, B), 128 threads = 4 waves.
// Each wave owns a 16-row q strip with full HD=64 accumulator and private
// online-softmax row stats (reductions stay inside a 16-lane half).
// ---------------------------------------------------------------------------
__global__ __launch_bounds__(128) void flash_attn_kernel(
    const half_t* __restrict__ qh, const half_t* __restrict__ kh,
    const half_t* __restrict__ vh, half_t* __restrict__ oh) {
  __shared__ alignas(16) half_t lK[64][72];        // K tile [key][d] (== B^T)
  __shared__ alignas(16) half_t lVT[64][72];       // V tile transposed [d][key]
  __shared__ alignas(16) half_t lP[4][16][64];     // per-wave P staging

  const int tid  = threadIdx.x;
  const int wave = tid >> 5;
  const int lane = tid & 31;
  const int hi   = lane >> 4;
  const int ln   = lane & 15;
  const int qt = blockIdx.x, h = blockIdx.y, b = blockIdx.z;

  const size_t bh_off = (size_t)b * SEQ * DMODEL + (size_t)h * HDIM;

  // Q fragments for this wave's 16-row strip (resident in VGPRs all kernel).
  const int qrow = qt * 64 + wave * 16 + ln;
  const half_t* qp = qh + bh_off + (size_t)qrow * DMODEL;
  const v16h qf0 = load_frag_a(qp, hi);        // d = 0..31
  const v16h qf1 = load_frag_a(qp + 32, hi);   // d = 32..63

  v8f o[4] = {{}, {}, {}, {}};
  float mrow[8], lrow[8];
#pragma unroll
  for (int r = 0; r < 8; ++r) { mrow[r] = -1e30f; lrow[r] = 0.f; }

  for (int kt = 0; kt < SEQ; kt += 64) {
    // Stage K tile row-major (this is exactly the transposed-B layout for QK^T)
    {
      const int key = tid >> 1, d0 = (tid & 1) * 32;
      const half_t* src = kh + bh_off + (size_t)(kt + key) * DMODEL + d0;
      half_t* dst = &lK[key][d0];
#pragma unroll
      for (int i = 0; i < 4; ++i) *(v8h*)(dst + i * 8) = *(const v8h*)(src + i * 8);
    }
    // Stage V transposed: lVT[d][key]
    {
      const int key = tid >> 1, d0 = (tid & 1) * 32;
      const half_t* src = vh + bh_off + (size_t)(kt + key) * DMODEL + d0;
#pragma unroll
      for (int i = 0; i < 32; ++i) lVT[d0 + i][key] = src[i];
    }
    if (kt + 64 < SEQ)  // gfx1250 global_prefetch for the next K tile
      __builtin_prefetch(kh + bh_off + (size_t)(kt + 64 + (tid >> 1)) * DMODEL, 0, 0);
    __syncthreads();

    // ---- scores S = (Q K^T) / sqrt(HD), 4 tiles of 16 keys ----
    v8f st[4];
#pragma unroll
    for (int j = 0; j < 4; ++j) {
      const half_t* kp = &lK[j * 16 + ln][0];
      v8f s = {};
      s = wmma32(qf0, load_frag_b(kp, hi), s);
      s = wmma32(qf1, load_frag_b(kp + 32, hi), s);
      st[j] = s * 0.125f;
    }

    // ---- online softmax (row = r + 8*hi, cols across 16-lane half) ----
    float mnew[8];
#pragma unroll
    for (int r = 0; r < 8; ++r) {
      float mx = fmaxf(fmaxf(st[0][r], st[1][r]), fmaxf(st[2][r], st[3][r]));
      mx = fmaxf(mx, __shfl_xor(mx, 1));
      mx = fmaxf(mx, __shfl_xor(mx, 2));
      mx = fmaxf(mx, __shfl_xor(mx, 4));
      mx = fmaxf(mx, __shfl_xor(mx, 8));
      mnew[r] = fmaxf(mrow[r], mx);
    }
#pragma unroll
    for (int r = 0; r < 8; ++r) {
      const float scale = __expf(mrow[r] - mnew[r]);
      mrow[r] = mnew[r];
      float psum = 0.f;
#pragma unroll
      for (int j = 0; j < 4; ++j) {
        const float p = __expf(st[j][r] - mnew[r]);
        st[j][r] = p;
        psum += p;
      }
      psum += __shfl_xor(psum, 1);
      psum += __shfl_xor(psum, 2);
      psum += __shfl_xor(psum, 4);
      psum += __shfl_xor(psum, 8);
      lrow[r] = lrow[r] * scale + psum;
#pragma unroll
      for (int j = 0; j < 4; ++j) o[j][r] = o[j][r] * scale;
    }

    // ---- C-layout P -> A-layout via wave-private LDS ----
#pragma unroll
    for (int j = 0; j < 4; ++j)
#pragma unroll
      for (int r = 0; r < 8; ++r)
        lP[wave][r + 8 * hi][j * 16 + ln] = (half_t)st[j][r];
    asm volatile("s_wait_dscnt 0" ::: "memory");  // same-wave LDS RAW fence

    const half_t* prow = &lP[wave][ln][0];
    const v16h p0 = load_frag_a(prow, hi);        // keys 0..31
    const v16h p1 = load_frag_a(prow + 32, hi);   // keys 32..63

    // ---- O += P V  (B = V^T stored [d][key]) ----
#pragma unroll
    for (int j = 0; j < 4; ++j) {
      const half_t* vp = &lVT[j * 16 + ln][0];
      o[j] = wmma32(p0, load_frag_b(vp, hi), o[j]);
      o[j] = wmma32(p1, load_frag_b(vp + 32, hi), o[j]);
    }
    __syncthreads();
  }

  // ---- normalize and write attn output (concat layout: col = h*64 + d) ----
#pragma unroll
  for (int r = 0; r < 8; ++r) {
    const float inv = 1.0f / lrow[r];
    const int row = qt * 64 + wave * 16 + r + 8 * hi;
    half_t* dst = oh + bh_off + (size_t)row * DMODEL;
#pragma unroll
    for (int j = 0; j < 4; ++j) dst[j * 16 + ln] = (half_t)(o[j][r] * inv);
  }
}

// ---------------------------------------------------------------------------
// Kernel 3: output projection  d_out[ROWS,256] = attn_f16 @ Wo + bo  (fp32)
// ---------------------------------------------------------------------------
__global__ __launch_bounds__(256) void out_proj_kernel(
    const half_t* __restrict__ A, const float* __restrict__ W,
    const float* __restrict__ bias, float* __restrict__ out) {
  constexpr int TM = 128, TN = 64, TK = 32;
  __shared__ alignas(16) half_t lA[TM][TK];
  __shared__ alignas(16) half_t lB[TN][TK];

  const int tid  = threadIdx.x;
  const int wave = tid >> 5;
  const int lane = tid & 31;
  const int hi   = lane >> 4;
  const int ln   = lane & 15;
  const int row0 = blockIdx.x * TM;
  const int n0   = blockIdx.y * TN;

  v8f acc[4] = {{}, {}, {}, {}};

  for (int kt = 0; kt < DMODEL; kt += TK) {
    {
      const int base = tid * 16;
      const int r = base >> 5, c = base & 31;  // c in {0,16} -> 16B aligned
      const half_t* src = A + (size_t)(row0 + r) * DMODEL + kt + c;
      *(v8h*)&lA[r][c]     = *(const v8h*)src;
      *(v8h*)&lA[r][c + 8] = *(const v8h*)(src + 8);
    }
    {
      const int c = tid >> 3;
      const int n = (tid & 7) * 8;
      const float* src = W + (size_t)(kt + c) * DMODEL + n0 + n;
#pragma unroll
      for (int i = 0; i < 8; ++i) lB[n + i][c] = (half_t)src[i];
    }
    __syncthreads();

    const v16h a = load_frag_a(&lA[wave * 16 + ln][0], hi);
#pragma unroll
    for (int j = 0; j < 4; ++j) {
      const v16h b = load_frag_b(&lB[j * 16 + ln][0], hi);
      acc[j] = wmma32(a, b, acc[j]);
    }
    __syncthreads();
  }

#pragma unroll
  for (int j = 0; j < 4; ++j) {
    const int col = n0 + j * 16 + ln;
    const float bv = bias[col];
#pragma unroll
    for (int r = 0; r < 8; ++r) {
      const int row = row0 + wave * 16 + r + 8 * hi;
      out[(size_t)row * DMODEL + col] = acc[j][r] + bv;
    }
  }
}

// ---------------------------------------------------------------------------
extern "C" void kernel_launch(void* const* d_in, const int* in_sizes, int n_in,
                              void* d_out, int out_size, void* d_ws, size_t ws_size,
                              hipStream_t stream) {
  (void)in_sizes; (void)n_in; (void)out_size; (void)ws_size;
  const float* x  = (const float*)d_in[0];
  const float* Wq = (const float*)d_in[1];
  const float* bq = (const float*)d_in[2];
  const float* Wk = (const float*)d_in[3];
  const float* bk = (const float*)d_in[4];
  const float* Wv = (const float*)d_in[5];
  const float* bv = (const float*)d_in[6];
  const float* Wo = (const float*)d_in[7];
  const float* bo = (const float*)d_in[8];

  const size_t planeElems = (size_t)ROWS * DMODEL;  // 4M f16 = 8 MiB each
  half_t* qh = (half_t*)d_ws;
  half_t* kh = qh + planeElems;
  half_t* vh = kh + planeElems;
  half_t* ah = vh + planeElems;   // total 32 MiB of workspace

  const dim3 gridG(ROWS / 128, DMODEL / 64);
  qkv_proj_kernel<<<gridG, 256, 0, stream>>>(x, Wq, bq, qh);
  qkv_proj_kernel<<<gridG, 256, 0, stream>>>(x, Wk, bk, kh);
  qkv_proj_kernel<<<gridG, 256, 0, stream>>>(x, Wv, bv, vh);

  flash_attn_kernel<<<dim3(SEQ / 64, NHEAD, BATCH), 128, 0, stream>>>(qh, kh, vh, ah);

  out_proj_kernel<<<gridG, 256, 0, stream>>>(ah, Wo, bo, (float*)d_out);
}
